// JITLayerNormGRU_11854109737082
// MI455X (gfx1250) — matrix operand order repaired and checked by reading, not compiled, not run on hardware
//
#include <hip/hip_runtime.h>
#include <hip/hip_bf16.h>

// ---------------------------------------------------------------------------
// LayerNorm-GRU for MI455X (gfx1250, wave32, WMMA + TDM).
//   B=64, T=512, D=512, H=512, 3H=1536
// Kernel 1: fp32 -> bf16 rounding of x, w_ih, w_hh (RNE).
// Kernel 2: x_gates = x @ w_ih^T + b_ih  via v_wmma_f32_16x16x32_bf16.
// Kernel 3: 4 persistent blocks (16 batch rows each, rows independent).
//           Per step: TDM async-stages the step's x_gates tile into LDS
//           (overlapped with the h-gates WMMA GEMM), then LN+sigmoid (r,z),
//           LN+tanh (n) with wave32 shuffle reductions; fp32 state in LDS.
// ---------------------------------------------------------------------------

#define B_  64
#define T_  512
#define D_  512
#define H_  512
#define G3_ 1536
#define BT_ (B_ * T_)

typedef __attribute__((ext_vector_type(16))) __bf16          v16bf;
typedef __attribute__((ext_vector_type(8)))  float           v8f;
typedef __attribute__((ext_vector_type(8)))  unsigned short  ushort8;
typedef __attribute__((ext_vector_type(16))) unsigned short  ushort16;
typedef __attribute__((ext_vector_type(4)))  unsigned int    uint4v;
typedef __attribute__((ext_vector_type(8)))  int             int8v;
typedef __attribute__((ext_vector_type(4)))  int             int4v;

union FragA {
    v16bf   bf;
    ushort8 h[2];
};
union FragB {
    v16bf    bf;
    ushort16 us;
};

__device__ __forceinline__ unsigned short f32_to_bf16_rne(float f) {
    unsigned int u = __float_as_uint(f);
    u += 0x7FFFu + ((u >> 16) & 1u);   // round-to-nearest-even
    return (unsigned short)(u >> 16);
}

// A fragment (16x32 bf16, MxK): lane l holds row m = l%16.
// half = l/16: VGPRs 0-3 hold K = kc + half*16 .. +7, VGPRs 4-7 hold K = kc+half*16+8 .. +7
__device__ __forceinline__ v16bf load_frag_a(const unsigned short* base, int ld,
                                             int m_base, int kc) {
    const int lane = threadIdx.x & 31;
    const int m    = m_base + (lane & 15);
    const int hsel = (lane >> 4) * 16;
    const unsigned short* p = base + (long)m * ld + kc + hsel;
    FragA f;
    f.h[0] = *(const ushort8*)(p);
    f.h[1] = *(const ushort8*)(p + 8);
    return f.bf;
}

// B fragment (32x16 bf16, KxN) sourced from row-major W[N,K] (so B[k,n]=W[n,k]):
// lane l holds column n = l%16, K = kc + (l/16)*16 .. +15 (one contiguous 32B read).
__device__ __forceinline__ v16bf load_frag_b(const unsigned short* base, int ld,
                                             int n_base, int kc) {
    const int lane = threadIdx.x & 31;
    const int n    = n_base + (lane & 15);
    const int koff = kc + ((lane >> 4) << 4);
    FragB f;
    f.us = *(const ushort16*)(base + (long)n * ld + koff);
    return f.bf;
}

__device__ __forceinline__ float wave_allreduce_sum(float v) {
    for (int off = 16; off > 0; off >>= 1) v += __shfl_xor(v, off, 32);
    return v;
}

// LDS byte layout for the scan kernel (dynamic LDS, 240 KB total)
#define LDS_HF_OFF   0                        // 16*512 f32   = 32 KB
#define LDS_HB_OFF   32768                    // 16*512 bf16  = 16 KB
#define LDS_HG_OFF   49152                    // 16*1536 f32  = 96 KB
#define LDS_XG_OFF   147456                   // 16*1536 f32  = 96 KB (TDM dest)
#define LDS_TOTAL    245760

// Issue a TDM tensor_load_to_lds copying the 16x1536 f32 x_gates tile for
// (rows row0..row0+15, timestep t) into LDS at LDS_XG_OFF.
// 2-D tensor: dim0 (contiguous) = 1536 f32, row stride = T*3H f32.
__device__ __forceinline__ void tdm_stage_xgates(const float* xg, int row0, int t) {
    const unsigned long long gaddr =
        (unsigned long long)(const void*)(xg + ((long)row0 * T_ + t) * G3_);

    uint4v g0;
    g0[0] = 1u;                                           // count=1 (valid user D#)
    g0[1] = (unsigned int)LDS_XG_OFF;                     // lds_addr (bytes)
    g0[2] = (unsigned int)(gaddr & 0xFFFFFFFFu);          // global_addr[31:0]
    g0[3] = (unsigned int)((gaddr >> 32) & 0x01FFFFFFu)   // global_addr[56:32]
            | 0x80000000u;                                // type = 2 ("image")

    const unsigned int dim0    = (unsigned int)G3_;       // tensor_dim0 = 1536
    const unsigned int dim1    = 16u;                     // tensor_dim1 = 16 rows
    const unsigned int stride0 = (unsigned int)(T_ * G3_);// row stride (elements)

    int8v g1;
    g1[0] = (int)(2u << 16);                              // data_size = 2 -> 4 bytes
    g1[1] = (int)((dim0 & 0xFFFFu) << 16);                // tensor_dim0[15:0]
    g1[2] = (int)(((dim0 >> 16) & 0xFFFFu)                // tensor_dim0[31:16]
            | ((dim1 & 0xFFFFu) << 16));                  // tensor_dim1[15:0]
    g1[3] = (int)(((dim1 >> 16) & 0xFFFFu)                // tensor_dim1[31:16]
            | ((unsigned int)G3_ << 16));                 // tile_dim0 = 1536
    g1[4] = 16;                                           // tile_dim1 = 16, tile_dim2 = 0
    g1[5] = (int)stride0;                                 // tensor_dim0_stride[31:0]
    g1[6] = 0;                                            // stride0 hi / stride1 lo
    g1[7] = 0;

    int4v g2 = {0, 0, 0, 0};
    int4v g3 = {0, 0, 0, 0};

#if defined(__clang_major__) && (__clang_major__ >= 23)
    int8v g4 = {0, 0, 0, 0, 0, 0, 0, 0};
    __builtin_amdgcn_tensor_load_to_lds(g0, g1, g2, g3, g4, 0);
#else
    __builtin_amdgcn_tensor_load_to_lds(g0, g1, g2, g3, 0);
#endif
}

// ---------------------------------------------------------------------------
// Kernel 1: fp32 -> bf16
// ---------------------------------------------------------------------------
__global__ void cvt_bf16_kernel(const float* __restrict__ in,
                                unsigned short* __restrict__ out, long n) {
    long i = (long)blockIdx.x * blockDim.x + threadIdx.x;
    long stride = (long)gridDim.x * blockDim.x;
    for (; i < n; i += stride) out[i] = f32_to_bf16_rne(in[i]);
}

// ---------------------------------------------------------------------------
// Kernel 2: x_gates[BT, 3H] = x_bf16[BT, D] @ w_ih_bf16[3H, D]^T + b_ih
// Block = 64x64 tile, 256 threads (8 waves). Wave w: M-subtile (w&3),
// two N-subtiles ((w>>2)*2 + {0,1}) -> A fragment reused across both WMMAs.
// ---------------------------------------------------------------------------
__global__ __launch_bounds__(256)
void xgates_gemm_kernel(const unsigned short* __restrict__ xb,
                        const unsigned short* __restrict__ wb,
                        const float* __restrict__ bias,
                        float* __restrict__ out) {
    const int bm   = blockIdx.x * 64;     // over BT
    const int bn   = blockIdx.y * 64;     // over 3H
    const int wave = threadIdx.x >> 5;
    const int lane = threadIdx.x & 31;
    const int mt   = wave & 3;
    const int nt0  = (wave >> 2) * 2;
    const int m_sub = bm + mt * 16;

    v8f c[2];
    for (int i = 0; i < 2; ++i) {
        const float b = bias[bn + (nt0 + i) * 16 + (lane & 15)];
        for (int r = 0; r < 8; ++r) c[i][r] = b;
    }

    for (int kc = 0; kc < D_; kc += 32) {
        if (kc + 32 < D_) {
            __builtin_prefetch(wb + (long)(bn + nt0 * 16 + (lane & 15)) * D_ + kc + 32, 0, 0);
        }
        const v16bf a = load_frag_a(xb, D_, m_sub, kc);
        for (int i = 0; i < 2; ++i) {
            const v16bf bf = load_frag_b(wb, D_, bn + (nt0 + i) * 16, kc);
            c[i] = __builtin_amdgcn_wmma_f32_16x16x32_bf16(
                       false, a, false, bf, (short)0, c[i], false, false);
        }
    }

    // C/D layout: VGPR r, lane l -> M = r + (l>>4)*8, N = l&15
    for (int i = 0; i < 2; ++i) {
        const int n  = bn + (nt0 + i) * 16 + (lane & 15);
        const int m0 = m_sub + (lane >> 4) * 8;
        for (int r = 0; r < 8; ++r)
            out[(long)(m0 + r) * G3_ + n] = c[i][r];
    }
}

// ---------------------------------------------------------------------------
// Kernel 3: persistent recurrent scan. grid=4, block=1024 (32 wave32s).
// Block b owns batch rows [16b, 16b+16). Dynamic LDS (240 KB):
//   h_f    : 16x512 f32   fp32 hidden state
//   h_b    : 16x512 bf16  WMMA A operand
//   hg     : 16x1536 f32  h_gates, later overwritten by r,z in-place
//   xg_lds : 16x1536 f32  x_gates tile, filled each step by the TDM
// ---------------------------------------------------------------------------
__global__ __launch_bounds__(1024)
void gru_scan_kernel(const float* __restrict__ h0,
                     const unsigned short* __restrict__ whh,
                     const float* __restrict__ b_hh,
                     const float* __restrict__ xg,
                     const float* __restrict__ ln_r_w, const float* __restrict__ ln_r_b,
                     const float* __restrict__ ln_z_w, const float* __restrict__ ln_z_b,
                     const float* __restrict__ ln_n_w, const float* __restrict__ ln_n_b,
                     float* __restrict__ out, float* __restrict__ hlast) {
    extern __shared__ char smem[];
    float*          h_f    = (float*)(smem + LDS_HF_OFF);
    unsigned short* h_b    = (unsigned short*)(smem + LDS_HB_OFF);
    float*          hg     = (float*)(smem + LDS_HG_OFF);
    float*          xg_lds = (float*)(smem + LDS_XG_OFF);

    const int tid  = threadIdx.x;
    const int wave = tid >> 5;
    const int lane = tid & 31;
    const int row0 = blockIdx.x * 16;

    // load initial hidden state
    for (int i = tid; i < 16 * H_; i += 1024)
        h_f[i] = h0[(long)(row0 + (i >> 9)) * H_ + (i & 511)];
    __syncthreads();

    for (int t = 0; t < T_; ++t) {
        // ---- 0. kick off TDM DMA of this step's x_gates tile (wave 0) ----
        if (wave == 0) tdm_stage_xgates(xg, row0, t);

        // ---- 1. round state to bf16 for WMMA ----
        for (int i = tid; i < 16 * H_; i += 1024)
            h_b[i] = f32_to_bf16_rne(h_f[i]);
        __syncthreads();

        // ---- 2. h_gates[16,1536] = h_b[16,512] @ whh[1536,512]^T + b_hh ----
        // (overlaps with the in-flight TDM copy)
        for (int nt = wave; nt < 96; nt += 32) {
            const int n_base = nt * 16;
            v8f c;
            {
                const float bb = b_hh[n_base + (lane & 15)];
                for (int r = 0; r < 8; ++r) c[r] = bb;
            }
            for (int kc = 0; kc < H_; kc += 32) {
                const v16bf a  = load_frag_a(h_b, H_, 0, kc);
                const v16bf bf = load_frag_b(whh, H_, n_base, kc);
                c = __builtin_amdgcn_wmma_f32_16x16x32_bf16(
                        false, a, false, bf, (short)0, c, false, false);
            }
            const int n  = n_base + (lane & 15);
            const int m0 = (lane >> 4) * 8;
            for (int r = 0; r < 8; ++r)
                hg[(m0 + r) * G3_ + n] = c[r];
        }

        // ---- wait for the TDM tile, then publish to the workgroup ----
        if (wave == 0) __builtin_amdgcn_s_wait_tensorcnt(0);
        __syncthreads();

        // ---- 3a. r (waves 0-15) and z (waves 16-31): LN + sigmoid ----
        {
            const int gate = wave >> 4;          // 0 = r, 1 = z
            const int row  = wave & 15;
            const float* lw = gate ? ln_z_w : ln_r_w;
            const float* lb = gate ? ln_z_b : ln_r_b;
            float pre[16];
            float s = 0.f;
            for (int j = 0; j < 16; ++j) {
                const int n = lane + j * 32;
                const float v = xg_lds[row * G3_ + gate * H_ + n]
                              + hg[row * G3_ + gate * H_ + n];
                pre[j] = v; s += v;
            }
            const float mean = wave_allreduce_sum(s) * (1.f / (float)H_);
            float vs = 0.f;
            for (int j = 0; j < 16; ++j) { const float d = pre[j] - mean; vs += d * d; }
            const float inv = rsqrtf(wave_allreduce_sum(vs) * (1.f / (float)H_) + 1e-5f);
            for (int j = 0; j < 16; ++j) {
                const int n = lane + j * 32;
                const float y = (pre[j] - mean) * inv * lw[n] + lb[n];
                hg[row * G3_ + gate * H_ + n] = 1.f / (1.f + __expf(-y));
            }
        }
        __syncthreads();

        // ---- 3b. n gate (LN + tanh) and state update: waves 0-15, one row each ----
        if (wave < 16) {
            const int row = wave;
            float pre[16];
            float s = 0.f;
            for (int j = 0; j < 16; ++j) {
                const int n = lane + j * 32;
                const float r = hg[row * G3_ + n];                   // r gate
                const float v = xg_lds[row * G3_ + 2 * H_ + n]
                              + r * hg[row * G3_ + 2 * H_ + n];
                pre[j] = v; s += v;
            }
            const float mean = wave_allreduce_sum(s) * (1.f / (float)H_);
            float vs = 0.f;
            for (int j = 0; j < 16; ++j) { const float d = pre[j] - mean; vs += d * d; }
            const float inv = rsqrtf(wave_allreduce_sum(vs) * (1.f / (float)H_) + 1e-5f);
            for (int j = 0; j < 16; ++j) {
                const int n = lane + j * 32;
                const float y  = (pre[j] - mean) * inv * ln_n_w[n] + ln_n_b[n];
                const float nn = tanhf(y);
                const float z  = hg[row * G3_ + H_ + n];             // z gate
                const float hn = (1.f - z) * nn + z * h_f[row * H_ + n];
                h_f[row * H_ + n] = hn;
                out[((long)(row0 + row) * T_ + t) * H_ + n] = hn;
            }
        }
        __syncthreads();
    }

    for (int i = tid; i < 16 * H_; i += 1024)
        hlast[(long)(row0 + (i >> 9)) * H_ + (i & 511)] = h_f[i];
}

// ---------------------------------------------------------------------------
// Host launcher
// ---------------------------------------------------------------------------
extern "C" void kernel_launch(void* const* d_in, const int* in_sizes, int n_in,
                              void* d_out, int out_size, void* d_ws, size_t ws_size,
                              hipStream_t stream) {
    const float* x     = (const float*)d_in[0];
    const float* h     = (const float*)d_in[1];
    const float* w_ih  = (const float*)d_in[2];
    const float* b_ih  = (const float*)d_in[3];
    const float* w_hh  = (const float*)d_in[4];
    const float* b_hh  = (const float*)d_in[5];
    const float* lnrw  = (const float*)d_in[6];
    const float* lnrb  = (const float*)d_in[7];
    const float* lnzw  = (const float*)d_in[8];
    const float* lnzb  = (const float*)d_in[9];
    const float* lnnw  = (const float*)d_in[10];
    const float* lnnb  = (const float*)d_in[11];

    // workspace layout
    unsigned short* wihb = (unsigned short*)d_ws;                       // 1536*512 bf16
    unsigned short* whhb = wihb + (size_t)G3_ * D_;                     // 1536*512 bf16
    unsigned short* xb   = whhb + (size_t)G3_ * H_;                     // 64*512*512 bf16
    float*          xg   = (float*)(xb + (size_t)B_ * T_ * D_);         // 32768*1536 f32

    float* out   = (float*)d_out;                                       // [B,T,H]
    float* hlast = out + (size_t)B_ * T_ * H_;                          // [B,H]

    // 1. round to bf16
    cvt_bf16_kernel<<<512, 256, 0, stream>>>(w_ih, wihb, (long)G3_ * D_);
    cvt_bf16_kernel<<<512, 256, 0, stream>>>(w_hh, whhb, (long)G3_ * H_);
    cvt_bf16_kernel<<<4096, 256, 0, stream>>>(x, xb, (long)B_ * T_ * D_);

    // 2. input projection GEMM (bandwidth-bound, fully parallel)
    dim3 g2(BT_ / 64, G3_ / 64);
    xgates_gemm_kernel<<<g2, 256, 0, stream>>>(xb, wihb, b_ih, xg);

    // 3. persistent recurrent scan: 4 blocks x 1024 threads, 240 KB LDS each
    gru_scan_kernel<<<4, 1024, LDS_TOTAL, stream>>>(
        h, whhb, b_hh, xg, lnrw, lnrb, lnzw, lnzb, lnnw, lnnb, out, hlast);
}
